// SAGELayer_10866267259419
// MI455X (gfx1250) — matrix operand order repaired and checked by reading, not compiled, not run on hardware
//
#include <hip/hip_runtime.h>
#include <hip/hip_bf16.h>

// ---------------------------------------------------------------------------
// SAGE layer for MI455X (gfx1250, wave32):
//   1) scatter:  nb[dst] += val * input[src]        (fp32 atomics, L2-resident)
//   2) wt_conv:  Wt[n][k] = bf16(weight[k][n])      (transpose + downconvert)
//   3) gemm:     out[:, 0:256]   = input @ W        (v_wmma_f32_16x16x32_bf16)
//   4) gemm:     out[:, 256:512] = nb    @ W
// ---------------------------------------------------------------------------

#define IN_F 256
#define OUT_F 256
#define OUT_STRIDE 512

typedef __attribute__((ext_vector_type(16))) __bf16 v16bf;
typedef __attribute__((ext_vector_type(8)))  float  v8f;

// ---------------------------------------------------------------------------
// Weight transpose + fp32 -> bf16:  Wt[n*256 + k] = bf16(W[k*256 + n])
// 256x256, one thread per element. Reads coalesced over n.
// ---------------------------------------------------------------------------
__global__ void sage_wt_kernel(const float* __restrict__ w,
                               __bf16* __restrict__ wt) {
    int idx = blockIdx.x * 256 + threadIdx.x;   // 65536 threads
    int k = idx >> 8;
    int n = idx & 255;
    wt[(size_t)n * IN_F + k] = (__bf16)w[(size_t)k * IN_F + n];
}

// ---------------------------------------------------------------------------
// COO scatter-add: one wave (32 lanes) per edge; each lane moves 8 floats.
// input rows (100 MB) and nb accumulator live in the 192 MB L2.
// ---------------------------------------------------------------------------
__global__ __launch_bounds__(256) void sage_scatter_kernel(
    const float* __restrict__ x,
    const int*   __restrict__ esrc,
    const int*   __restrict__ edst,
    const float* __restrict__ eval,
    float* __restrict__ nb,
    int nE)
{
    int e = blockIdx.x * 8 + (threadIdx.x >> 5);
    if (e >= nE) return;
    int lane = threadIdx.x & 31;

    int   s = esrc[e];
    int   d = edst[e];
    float v = eval[e];

    const float4* xs = (const float4*)(x + (size_t)s * IN_F) + lane * 2;
    float4 a = xs[0];
    float4 b = xs[1];

    float* o = nb + (size_t)d * IN_F + lane * 8;
    atomicAdd(o + 0, v * a.x);
    atomicAdd(o + 1, v * a.y);
    atomicAdd(o + 2, v * a.z);
    atomicAdd(o + 3, v * a.w);
    atomicAdd(o + 4, v * b.x);
    atomicAdd(o + 5, v * b.y);
    atomicAdd(o + 6, v * b.z);
    atomicAdd(o + 7, v * b.w);
}

// ---------------------------------------------------------------------------
// GEMM: out[m, colOff + n] = sum_k src[m,k] * W[k,n], K = N_cols = 256.
// Block = 256 threads = 8 waves laid out 2(M) x 4(N): 32x256 tile per block.
// Each wave: 16 rows x 64 cols = 4 accumulators; K-loop of 8 x (16x16x32).
// A: fp32 rows converted to bf16 in-register. B: pre-transposed bf16 Wt,
// one aligned 32B load per fragment.
// ---------------------------------------------------------------------------
__global__ __launch_bounds__(256, 1) void sage_gemm_kernel(
    const float*  __restrict__ src,
    const __bf16* __restrict__ wt,     // [N=256][K=256] bf16, row-major in K
    float* __restrict__ out,
    int nrows,
    int colOff)
{
    const int tid   = threadIdx.x;
    const int wave  = tid >> 5;
    const int lane  = tid & 31;
    const int waveM = wave >> 2;       // 0..1
    const int waveN = wave & 3;        // 0..3
    const int lrow  = lane & 15;
    const int hi    = lane >> 4;       // 0 or 1

    const int r0 = blockIdx.x * 32 + waveM * 16;
    const int n0 = waveN * 64;

    int m = r0 + lrow;
    if (m >= nrows) m = nrows - 1;     // clamp loads; stores guarded below
    const float* arow = src + (size_t)m * IN_F;

    v8f acc0 = {}, acc1 = {}, acc2 = {}, acc3 = {};

    for (int kk = 0; kk < IN_F; kk += 32) {
        if (kk + 32 < IN_F)
            __builtin_prefetch((const void*)(arow + kk + 32), 0, 1);

        // ---- A fragment (16x32 bf16): lane holds row lrow,
        //      K = ka..ka+15 contiguous, ka = kk + hi*16
        const int ka = kk + hi * 16;
        float4 a0 = *(const float4*)(arow + ka);
        float4 a1 = *(const float4*)(arow + ka + 4);
        float4 a2 = *(const float4*)(arow + ka + 8);
        float4 a3 = *(const float4*)(arow + ka + 12);
        v16bf af;
        af[0]  = (__bf16)a0.x; af[1]  = (__bf16)a0.y;
        af[2]  = (__bf16)a0.z; af[3]  = (__bf16)a0.w;
        af[4]  = (__bf16)a1.x; af[5]  = (__bf16)a1.y;
        af[6]  = (__bf16)a1.z; af[7]  = (__bf16)a1.w;
        af[8]  = (__bf16)a2.x; af[9]  = (__bf16)a2.y;
        af[10] = (__bf16)a2.z; af[11] = (__bf16)a2.w;
        af[12] = (__bf16)a3.x; af[13] = (__bf16)a3.y;
        af[14] = (__bf16)a3.z; af[15] = (__bf16)a3.w;

        // ---- B fragments (32x16 bf16): lane holds col n0+t*16+lrow,
        //      K = kb..kb+15 contiguous in Wt, kb = kk + hi*16
        const int kb = kk + hi * 16;
        const __bf16* bbase = wt + (size_t)(n0 + lrow) * IN_F + kb;
        v16bf b0 = *(const v16bf*)(bbase);
        v16bf b1 = *(const v16bf*)(bbase + 16 * IN_F);
        v16bf b2 = *(const v16bf*)(bbase + 32 * IN_F);
        v16bf b3 = *(const v16bf*)(bbase + 48 * IN_F);

        acc0 = __builtin_amdgcn_wmma_f32_16x16x32_bf16(false, af, false, b0,
                                                       (short)0, acc0, false, false);
        acc1 = __builtin_amdgcn_wmma_f32_16x16x32_bf16(false, af, false, b1,
                                                       (short)0, acc1, false, false);
        acc2 = __builtin_amdgcn_wmma_f32_16x16x32_bf16(false, af, false, b2,
                                                       (short)0, acc2, false, false);
        acc3 = __builtin_amdgcn_wmma_f32_16x16x32_bf16(false, af, false, b3,
                                                       (short)0, acc3, false, false);
    }

    // ---- store: VGPR r -> row r0 + r + hi*8, col n0 + t*16 + lrow
    const int srow = r0 + hi * 8;
    const int scol = colOff + n0 + lrow;
    #pragma unroll
    for (int r = 0; r < 8; ++r) {
        int row = srow + r;
        if (row < nrows) {
            float* orow = out + (size_t)row * OUT_STRIDE + scol;
            orow[0]  = acc0[r];
            orow[16] = acc1[r];
            orow[32] = acc2[r];
            orow[48] = acc3[r];
        }
    }
}

// ---------------------------------------------------------------------------
extern "C" void kernel_launch(void* const* d_in, const int* in_sizes, int n_in,
                              void* d_out, int out_size, void* d_ws, size_t ws_size,
                              hipStream_t stream) {
    const float* x    = (const float*)d_in[0];   // [N, 256] fp32
    const int*   esrc = (const int*)d_in[1];     // [E]
    const int*   edst = (const int*)d_in[2];     // [E]
    const float* ev   = (const float*)d_in[3];   // [E] fp32
    const float* w    = (const float*)d_in[4];   // [256, 256] fp32
    float* out = (float*)d_out;                  // [N, 512] fp32

    const int nNodes = in_sizes[0] / IN_F;
    const int nE     = in_sizes[1];

    // workspace: Wt bf16 (128 KB) | nb fp32 accumulator (nNodes*256*4 B)
    __bf16* wt = (__bf16*)d_ws;
    float*  nb = (float*)((char*)d_ws + 131072);

    hipMemsetAsync(nb, 0, (size_t)nNodes * IN_F * sizeof(float), stream);

    sage_wt_kernel<<<256, 256, 0, stream>>>(w, wt);

    sage_scatter_kernel<<<(nE + 7) / 8, 256, 0, stream>>>(x, esrc, edst, ev, nb, nE);

    const int mtiles = (nNodes + 31) / 32;
    sage_gemm_kernel<<<mtiles, 256, 0, stream>>>(x,  wt, out, nNodes, 0);
    sage_gemm_kernel<<<mtiles, 256, 0, stream>>>(nb, wt, out, nNodes, OUT_F);
}
